// CostVolumeLayer_17815524344422
// MI455X (gfx1250) — matrix operand order repaired and checked by reading, not compiled, not run on hardware
//
#include <hip/hip_runtime.h>

// Problem constants (from the reference)
#define BB   4
#define CC   128
#define HH   64
#define WW   128
#define RR   4
#define DD   9          // window diameter, DD*DD = 81 output channels
#define LDA  136        // LDS row stride in bf16 halves (128 + 8 pad, keeps 16B alignment)
#define NWAVE 3         // waves per workgroup, each owns 3 of the 9 di values

typedef __attribute__((ext_vector_type(16))) __bf16 v16bf;
typedef __attribute__((ext_vector_type(8)))  float  v8f;

union Frag { v16bf v; uint4 q[2]; };

// Convert two fp32 (adjacent K/channel values) into one packed bf16 dword.
#if __has_builtin(__builtin_amdgcn_cvt_pk_bf16_f32)
typedef __attribute__((ext_vector_type(2))) __bf16 v2bf;
__device__ __forceinline__ unsigned pack2bf(float a, float b) {
  v2bf p = __builtin_amdgcn_cvt_pk_bf16_f32(a, b);   // one v_cvt_pk_bf16_f32
  unsigned u;
  __builtin_memcpy(&u, &p, 4);
  return u;
}
#else
__device__ __forceinline__ unsigned pack2bf(float a, float b) {
  union { __bf16 h[2]; unsigned u; } x;
  x.h[0] = (__bf16)a;
  x.h[1] = (__bf16)b;
  return x.u;
}
#endif

__global__ __launch_bounds__(32 * NWAVE)
void costvol_wmma_kernel(const float* __restrict__ src,
                         const float* __restrict__ tgt,
                         float* __restrict__ out) {
  __shared__ __align__(16) unsigned short lds_a[16 * LDA];          // shared src tile [w][c]
  __shared__ __align__(16) unsigned short lds_b[NWAVE][32 * LDA];   // per-wave tgt row [p][c]
  __shared__ __align__(16) float          lds_e[NWAVE][32][16];     // per-wave diag-major C

  const int tid  = threadIdx.x;
  const int lane = tid & 31;
  const int wv   = tid >> 5;             // wave id 0..2
  const int w0   = blockIdx.x * 16;      // pixel tile base (M dim)
  const int h    = blockIdx.y;
  const int b    = blockIdx.z;

  // ---- Cooperative src stage: 256 tasks = 64 channel-pairs x 4 w-quads, stride 96.
  // Rounds split per-wave all-or-nothing, so EXEC never diverges inside a wave.
  for (int t = tid; t < 256; t += 32 * NWAVE) {
    const int c  = (t >> 2) * 2;         // even channel
    const int i4 = (t & 3) * 4;          // w quad
    const float4 fa = *reinterpret_cast<const float4*>(
        src + ((((size_t)b * CC + c) * HH + h) * WW + w0 + i4));
    const float4 fb = *reinterpret_cast<const float4*>(
        src + ((((size_t)b * CC + c + 1) * HH + h) * WW + w0 + i4));
    *reinterpret_cast<unsigned*>(&lds_a[(i4 + 0) * LDA + c]) = pack2bf(fa.x, fb.x);
    *reinterpret_cast<unsigned*>(&lds_a[(i4 + 1) * LDA + c]) = pack2bf(fa.y, fb.y);
    *reinterpret_cast<unsigned*>(&lds_a[(i4 + 2) * LDA + c]) = pack2bf(fa.z, fb.z);
    *reinterpret_cast<unsigned*>(&lds_a[(i4 + 3) * LDA + c]) = pack2bf(fa.w, fb.w);
  }
  __syncthreads();   // only cross-wave barrier: everything below is wave-private

  // ---- Build the 4 A fragments (K=128 in chunks of 32), per ISA 16-bit A layout:
  // lane<16 : M=lane,    K = {0..7}(q0) and {16..23}(q1) of each chunk
  // lane>=16: M=lane-16, K = {8..15}(q0) and {24..31}(q1)
  Frag afrag[4];
  {
    const unsigned short* abase = lds_a + (lane & 15) * LDA + ((lane >> 4) * 8);
#pragma unroll
    for (int k = 0; k < 4; ++k) {
      afrag[k].q[0] = *reinterpret_cast<const uint4*>(abase + 32 * k);
      afrag[k].q[1] = *reinterpret_cast<const uint4*>(abase + 32 * k + 16);
    }
  }

  unsigned short* const myb = lds_b[wv];
  float (* const mye)[16]   = lds_e[wv];

  // ---- Di-invariant extraction constants: 36 float4 quads per di, two uniform
  // rounds of 32 lanes (round 2 clamps to quad 35 -> benign duplicate stores).
  const int q1  = (32 + lane < 36) ? (32 + lane) : 35;
  const int dj0 = lane >> 2, mq0 = (lane & 3) * 4;
  const int dj1 = q1 >> 2,   mq1 = (q1 & 3) * 4;
  const size_t loff0 = (size_t)dj0 * (HH * WW) + mq0;
  const size_t loff1 = (size_t)dj1 * (HH * WW) + mq1;
  float* const obase = out + (((size_t)b * (DD * DD)) * HH + h) * WW + w0;

#pragma unroll 1
  for (int j = 0; j < DD / NWAVE; ++j) {
    const int  di    = wv * (DD / NWAVE) + j;   // this wave's displacement row
    const int  hr    = h - RR + di;
    const bool rowok = (hr >= 0) && (hr < HH);

    // ---- Prefetch this wave's next tgt row (gfx1250 global_prefetch_b8).
    {
      const int hr2 = hr + 1;
      if ((j + 1 < DD / NWAVE) && (hr2 >= 0) && (hr2 < HH)) {
        const int pw = (w0 >= RR) ? (w0 - RR) : 0;
        __builtin_prefetch(
            tgt + ((((size_t)b * CC + lane) * HH + hr2) * WW + pw), 0, 3);
      }
    }

    // ---- Stage shifted tgt row: 32 positions (w0-4 .. w0+27) x 128 c, zero-filled OOB.
    // Quads are 4-aligned and W%4==0 so each float4 is fully in- or out-of-range.
    {
      const int  n4  = (lane & 7) * 4;
      const int  c0  = (lane >> 3) * 2;  // even channel base 0,2,4,6
      const int  p   = w0 - RR + n4;
      const bool pok = rowok && (p >= 0) && (p + 3 < WW);
#pragma unroll 4
      for (int it = 0; it < 16; ++it) {
        const int c = c0 + it * 8;
        float4 fa = make_float4(0.f, 0.f, 0.f, 0.f);
        float4 fb = fa;
        if (pok) {
          fa = *reinterpret_cast<const float4*>(
              tgt + ((((size_t)b * CC + c) * HH + hr) * WW + p));
          fb = *reinterpret_cast<const float4*>(
              tgt + ((((size_t)b * CC + c + 1) * HH + hr) * WW + p));
        }
        *reinterpret_cast<unsigned*>(&myb[(n4 + 0) * LDA + c]) = pack2bf(fa.x, fb.x);
        *reinterpret_cast<unsigned*>(&myb[(n4 + 1) * LDA + c]) = pack2bf(fa.y, fb.y);
        *reinterpret_cast<unsigned*>(&myb[(n4 + 2) * LDA + c]) = pack2bf(fa.z, fb.z);
        *reinterpret_cast<unsigned*>(&myb[(n4 + 3) * LDA + c]) = pack2bf(fa.w, fb.w);
      }
    }
    // No barrier needed: LDS ops are in-order within a wave, buffers are wave-private.

    // ---- 2 N-tiles x 4 K-chunks of v_wmma_f32_16x16x32_bf16 (EXEC uniform here).
    // B layout (matches A): lane<16 : N=lane, K={0..7},{16..23}; lane>=16: K={8..15},{24..31}.
    v8f acc0 = {};
    v8f acc1 = {};
    const unsigned short* bbase0 = myb + (lane & 15) * LDA + ((lane >> 4) * 8);
    const unsigned short* bbase1 = bbase0 + 16 * LDA;   // second N-tile (positions w0+12..)
#pragma unroll
    for (int k = 0; k < 4; ++k) {
      Frag bf0, bf1;
      bf0.q[0] = *reinterpret_cast<const uint4*>(bbase0 + 32 * k);
      bf0.q[1] = *reinterpret_cast<const uint4*>(bbase0 + 32 * k + 16);
      bf1.q[0] = *reinterpret_cast<const uint4*>(bbase1 + 32 * k);
      bf1.q[1] = *reinterpret_cast<const uint4*>(bbase1 + 32 * k + 16);
      acc0 = __builtin_amdgcn_wmma_f32_16x16x32_bf16(
          false, afrag[k].v, false, bf0.v, (short)0, acc0, false, false);
      acc1 = __builtin_amdgcn_wmma_f32_16x16x32_bf16(
          false, afrag[k].v, false, bf1.v, (short)0, acc1, false, false);
    }

    // ---- Spill C tiles diagonal-major: element (M, n) -> mye[(n-M)&31][M].
    // C VGPR layout: VGPR r, lane l -> M = r + 8*(l>=16), n = (l&15) + 16*tile.
    {
      const int ncol = lane & 15;
      const int rofs = (lane >> 4) * 8;
#pragma unroll
      for (int r = 0; r < 8; ++r) {
        const int M = r + rofs;
        mye[(ncol - M) & 31][M]      = acc0[r];   // tile0: n = ncol
        mye[(ncol + 16 - M) & 31][M] = acc1[r];   // tile1: n = ncol + 16
      }
    }

    // ---- Write the 9 diagonals: out[b, di*9+dj, h, w0+mq..mq+3] = mye[dj][mq..mq+3]
    {
      float* const od = obase + (size_t)di * (DD * HH * WW);
      const float4 v0 = *reinterpret_cast<const float4*>(&mye[dj0][mq0]);
      const float4 v1 = *reinterpret_cast<const float4*>(&mye[dj1][mq1]);
      *reinterpret_cast<float4*>(od + loff0) = v0;
      *reinterpret_cast<float4*>(od + loff1) = v1;
    }
  }
}

extern "C" void kernel_launch(void* const* d_in, const int* in_sizes, int n_in,
                              void* d_out, int out_size, void* d_ws, size_t ws_size,
                              hipStream_t stream) {
  const float* src = (const float*)d_in[0];
  const float* tgt = (const float*)d_in[1];
  float* out = (float*)d_out;

  dim3 grid(WW / 16, HH, BB);       // 8 x 64 x 4 = 2048 workgroups
  dim3 block(32 * NWAVE);           // 3 waves/WG, one di-triple per wave -> 6144 waves
  costvol_wmma_kernel<<<grid, block, 0, stream>>>(src, tgt, out);
}
